// OwnGCN_73443940761887
// MI455X (gfx1250) — compile-verified, matrix operand order checked by reading, not run on hardware
//
#include <hip/hip_runtime.h>
#include <hip/hip_bf16.h>

typedef __attribute__((ext_vector_type(16))) __bf16 v16bf;
typedef __attribute__((ext_vector_type(8)))  __bf16 v8bf;
typedef __attribute__((ext_vector_type(8)))  float  v8f;

#define TPB 256

// ---------------- elementwise / setup kernels ----------------

__global__ void k_zero(float* p, long n) {
    long i = (long)blockIdx.x * blockDim.x + threadIdx.x;
    if (i < n) p[i] = 0.f;
}

__global__ void k_copy_scale(float* __restrict__ y, const float* __restrict__ x,
                             float a, long n) {
    long i = (long)blockIdx.x * blockDim.x + threadIdx.x;
    if (i < n) y[i] = a * x[i];
}

__global__ void k_deg(const int* __restrict__ src, const int* __restrict__ dst,
                      float* deg, int E) {
    int e = blockIdx.x * blockDim.x + threadIdx.x;
    if (e < E) {
        if (src[e] != dst[e]) atomicAdd(&deg[src[e]], 1.f);
    }
}

__global__ void k_dis(const float* __restrict__ deg, float* __restrict__ dis, int n) {
    int i = blockIdx.x * blockDim.x + threadIdx.x;
    if (i < n) {
        float d = deg[i];
        dis[i] = (d > 0.f) ? rsqrtf(d) : 0.f;
    }
}

__global__ void k_norm(const int* __restrict__ src, const int* __restrict__ dst,
                       const float* __restrict__ dis, float* __restrict__ nrm, int E) {
    int e = blockIdx.x * blockDim.x + threadIdx.x;
    if (e < E) {
        int s = src[e], d = dst[e];
        nrm[e] = (s != d) ? (-dis[s] * dis[d]) : 0.f;
    }
}

// out[dst] += factor * norm[e] * h[src]  (atomic scatter through L2)
__global__ void k_spmm(const int* __restrict__ src, const int* __restrict__ dst,
                       const float* __restrict__ nrm, const float* __restrict__ h,
                       float* __restrict__ out, int E, int F, float factor) {
    int chunks = (F + 3) >> 2;
    long tid = (long)blockIdx.x * blockDim.x + threadIdx.x;
    long tot = (long)E * chunks;
    if (tid >= tot) return;
    int e = (int)(tid / chunks);
    int f0 = (int)(tid % chunks) * 4;
    float w = factor * nrm[e];
    if (w == 0.f) return;
    long sb = (long)src[e] * F;
    long db = (long)dst[e] * F;
    #pragma unroll
    for (int j = 0; j < 4; ++j) {
        int f = f0 + j;
        if (f < F) atomicAdd(&out[db + f], w * h[sb + f]);
    }
}

// ---------------- bf16 staging (pad K to 32, pad rows; transpose W) ----------------

// dst[Mpad x Kpad] bf16 <- src[M x K] f32 (row-major), zero padding
__global__ void k_cvt_a(__bf16* __restrict__ dst, const float* __restrict__ src,
                        int M, int K, int Kpad, long tot) {
    long i = (long)blockIdx.x * blockDim.x + threadIdx.x;
    if (i >= tot) return;
    int r = (int)(i / Kpad), k = (int)(i % Kpad);
    float v = (r < M && k < K) ? src[(long)r * K + k] : 0.f;
    dst[i] = (__bf16)v;
}

// dst[Ncpad x Kpad] bf16 <- transpose of src[K x Nc] f32, zero padding
__global__ void k_cvt_wt(__bf16* __restrict__ dst, const float* __restrict__ src,
                         int K, int Nc, int Kpad, long tot) {
    long i = (long)blockIdx.x * blockDim.x + threadIdx.x;
    if (i >= tot) return;
    int n = (int)(i / Kpad), k = (int)(i % Kpad);
    float v = (n < Nc && k < K) ? src[(long)k * Nc + n] : 0.f;
    dst[i] = (__bf16)v;
}

// ---------------- WMMA bf16 GEMM: C (+)= Abf[MxKpad] @ Wt^T ----------------
// One wave computes one 16x16 output tile. Operands are pre-staged bf16 with
// Kpad % 32 == 0, so the K loop is pure b128 loads + v_wmma (no branches).
__global__ void __launch_bounds__(TPB)
k_gemm_wmma(const __bf16* __restrict__ Abf, const __bf16* __restrict__ Wt,
            float* __restrict__ C, int M, int Kpad, int Nc, int accum) {
    int wave   = threadIdx.x >> 5;
    int lane   = threadIdx.x & 31;
    int half   = lane >> 4;    // lane group (0: lanes 0-15, 1: lanes 16-31)
    int mr     = lane & 15;    // row (A) / col (B,C) within the tile
    int tilesN = (Nc + 15) >> 4;
    int tilesM = (M + 15) >> 4;
    int tile   = blockIdx.x * 8 + wave;            // uniform per wave
    if (tile >= tilesM * tilesN) return;           // whole-wave exit: EXEC all-1s for WMMA
    int tm = tile / tilesN;
    int tn = tile % tilesN;

    v8f acc;
    #pragma unroll
    for (int r = 0; r < 8; ++r) {
        int row = tm * 16 + r + 8 * half;
        int col = tn * 16 + mr;
        float cv = 0.f;
        if (accum && row < M && col < Nc) cv = C[(long)row * Nc + col];
        acc[r] = cv;
    }

    // Per-lane contiguous runs of 8 bf16: A lane holds row tm*16+mr,
    // K = 32s + 8*half + [0,8) and 32s + 16 + 8*half + [0,8).
    const v8bf* Ar = (const v8bf*)(Abf + (size_t)(tm * 16 + mr) * Kpad);
    const v8bf* Br = (const v8bf*)(Wt  + (size_t)(tn * 16 + mr) * Kpad);
    int steps = Kpad >> 5;
    #pragma unroll 2
    for (int s = 0; s < steps; ++s) {
        v8bf alo = Ar[4 * s + half];
        v8bf ahi = Ar[4 * s + 2 + half];
        v8bf blo = Br[4 * s + half];
        v8bf bhi = Br[4 * s + 2 + half];
        v16bf av = __builtin_shufflevector(alo, ahi, 0, 1, 2, 3, 4, 5, 6, 7,
                                           8, 9, 10, 11, 12, 13, 14, 15);
        v16bf bv = __builtin_shufflevector(blo, bhi, 0, 1, 2, 3, 4, 5, 6, 7,
                                           8, 9, 10, 11, 12, 13, 14, 15);
        acc = __builtin_amdgcn_wmma_f32_16x16x32_bf16(
                  false, av, false, bv, (short)0, acc, false, false);
    }

    #pragma unroll
    for (int r = 0; r < 8; ++r) {
        int row = tm * 16 + r + 8 * half;
        int col = tn * 16 + mr;
        if (row < M && col < Nc) C[(long)row * Nc + col] = acc[r];
    }
}

// ---------------- epilogues ----------------

// act: 0 = none, 1 = relu, 2 = leaky relu(0.2)
__global__ void k_bias_act(float* __restrict__ x, const float* __restrict__ b,
                           long M, int C, int act) {
    long i = (long)blockIdx.x * blockDim.x + threadIdx.x;
    long tot = M * C;
    if (i >= tot) return;
    int c = (int)(i % C);
    float v = x[i] + b[c];
    if (act == 1) v = fmaxf(v, 0.f);
    else if (act == 2) v = (v > 0.f) ? v : 0.2f * v;
    x[i] = v;
}

__global__ void k_colmean(const float* __restrict__ x, float* __restrict__ m,
                          int M, int C) {
    __shared__ float s[TPB];
    int c = blockIdx.x;
    float acc = 0.f;
    for (int i = threadIdx.x; i < M; i += TPB) acc += x[(long)i * C + c];
    s[threadIdx.x] = acc;
    __syncthreads();
    for (int st = TPB / 2; st > 0; st >>= 1) {
        if (threadIdx.x < st) s[threadIdx.x] += s[threadIdx.x + st];
        __syncthreads();
    }
    if (threadIdx.x == 0) m[c] = s[0] / (float)M;
}

__global__ void k_colvar(const float* __restrict__ x, const float* __restrict__ m,
                         const float* __restrict__ ms, float* __restrict__ v,
                         int M, int C) {
    __shared__ float s[TPB];
    int c = blockIdx.x;
    float mm = m[c] * ms[c];
    float acc = 0.f;
    for (int i = threadIdx.x; i < M; i += TPB) {
        float o = x[(long)i * C + c] - mm;
        acc += o * o;
    }
    s[threadIdx.x] = acc;
    __syncthreads();
    for (int st = TPB / 2; st > 0; st >>= 1) {
        if (threadIdx.x < st) s[threadIdx.x] += s[threadIdx.x + st];
        __syncthreads();
    }
    if (threadIdx.x == 0) v[c] = s[0] / (float)M;
}

__global__ void k_gn_act(float* __restrict__ x, const float* __restrict__ m,
                         const float* __restrict__ ms, const float* __restrict__ v,
                         const float* __restrict__ w, const float* __restrict__ b,
                         long M, int C, int act) {
    long i = (long)blockIdx.x * blockDim.x + threadIdx.x;
    long tot = M * C;
    if (i >= tot) return;
    int c = (int)(i % C);
    float o = x[i] - m[c] * ms[c];
    float y = w[c] * o * rsqrtf(v[c] + 1e-5f) + b[c];
    if (act == 2) y = (y > 0.f) ? y : 0.2f * y;
    x[i] = y;
}

__global__ void k_pool(const float* __restrict__ h, const int* __restrict__ batch,
                       float* __restrict__ pooled, float* __restrict__ cnt,
                       int N, int C) {
    long i = (long)blockIdx.x * blockDim.x + threadIdx.x;
    long tot = (long)N * C;
    if (i >= tot) return;
    int n = (int)(i / C), c = (int)(i % C);
    int g = batch[n];
    atomicAdd(&pooled[(long)g * C + c], h[i]);
    if (c == 0) atomicAdd(&cnt[g], 1.f);
}

__global__ void k_pooldiv(float* pooled, const float* cnt, int G, int C) {
    int i = blockIdx.x * blockDim.x + threadIdx.x;
    if (i >= G * C) return;
    pooled[i] /= fmaxf(cnt[i / C], 1.f);
}

// act: 0 = none, 3 = tanh
__global__ void k_lin(const float* __restrict__ x, const float* __restrict__ W,
                      const float* __restrict__ b, float* __restrict__ y,
                      int M, int K, int Nc, int act) {
    int i = blockIdx.x * blockDim.x + threadIdx.x;
    if (i >= M * Nc) return;
    int r = i / Nc, c = i % Nc;
    float acc = b[c];
    for (int k = 0; k < K; ++k) acc += x[r * K + k] * W[k * Nc + c];
    if (act == 3) acc = tanhf(acc);
    y[i] = acc;
}

// ---------------- host orchestration ----------------

static inline unsigned gdim(long n) { return (unsigned)((n + TPB - 1) / TPB); }

extern "C" void kernel_launch(void* const* d_in, const int* in_sizes, int n_in,
                              void* d_out, int out_size, void* d_ws, size_t ws_size,
                              hipStream_t stream) {
    const float* x     = (const float*)d_in[0];
    const int*   ei    = (const int*)d_in[1];
    const int*   batch = (const int*)d_in[2];
    const float* convw[6] = {(const float*)d_in[3], (const float*)d_in[5], (const float*)d_in[7],
                             (const float*)d_in[9], (const float*)d_in[11], (const float*)d_in[13]};
    const float* convb[6] = {(const float*)d_in[4], (const float*)d_in[6], (const float*)d_in[8],
                             (const float*)d_in[10], (const float*)d_in[12], (const float*)d_in[14]};
    const float* bnw[4]  = {(const float*)d_in[15], (const float*)d_in[18], (const float*)d_in[21], (const float*)d_in[24]};
    const float* bnb[4]  = {(const float*)d_in[16], (const float*)d_in[19], (const float*)d_in[22], (const float*)d_in[25]};
    const float* bnms[4] = {(const float*)d_in[17], (const float*)d_in[20], (const float*)d_in[23], (const float*)d_in[26]};
    const float* lin1w = (const float*)d_in[27];
    const float* lin1b = (const float*)d_in[28];
    const float* lin2w = (const float*)d_in[29];
    const float* lin2b = (const float*)d_in[30];
    float* out_f = (float*)d_out;

    const int N = in_sizes[0] / 128;
    const int E = in_sizes[1] / 2;
    const int G = out_size / 10;
    const int* src = ei;
    const int* dst = ei + E;

    // workspace bump allocator
    char* ws = (char*)d_ws;
    size_t off = 0;
    auto alloc = [&](size_t bytes) -> void* {
        void* p = (void*)(ws + off);
        off += (bytes + 255) & ~(size_t)255;
        return p;
    };
    const int Npad16 = ((N + 15) / 16) * 16;
    const size_t big = (size_t)N * 256 * sizeof(float);
    float* bufs[4] = {(float*)alloc(big), (float*)alloc(big), (float*)alloc(big), (float*)alloc(big)};
    __bf16* Abf   = (__bf16*)alloc((size_t)Npad16 * 256 * sizeof(__bf16));
    __bf16* Wtbf  = (__bf16*)alloc((size_t)256 * 256 * sizeof(__bf16));
    float* deg    = (float*)alloc((size_t)N * 4);
    float* dis    = (float*)alloc((size_t)N * 4);
    float* nrm    = (float*)alloc((size_t)E * 4);
    float* mbuf   = (float*)alloc(1024);
    float* vbuf   = (float*)alloc(1024);
    float* pooled = (float*)alloc((size_t)G * 128 * 4);
    float* cnt    = (float*)alloc((size_t)G * 4);
    float* l1out  = (float*)alloc((size_t)G * 64 * 4);

    // ---- cheb norm ----
    k_zero<<<gdim(N), TPB, 0, stream>>>(deg, N);
    k_deg<<<gdim(E), TPB, 0, stream>>>(src, dst, deg, E);
    k_dis<<<gdim(N), TPB, 0, stream>>>(deg, dis, N);
    k_norm<<<gdim(E), TPB, 0, stream>>>(src, dst, dis, nrm, E);

    // C (+)= A[M x K] @ W[K x Nc], staging A and W^T as padded bf16 first
    auto gemm = [&](const float* A, const float* Wm, float* C, int M, int Kd, int Nc, int accum) {
        int Kpad  = (Kd + 31) & ~31;
        int Mpad  = ((M + 15) / 16) * 16;
        int Ncpad = ((Nc + 15) / 16) * 16;
        long at = (long)Mpad * Kpad;
        long wt = (long)Ncpad * Kpad;
        k_cvt_a<<<gdim(at), TPB, 0, stream>>>(Abf, A, M, Kd, Kpad, at);
        k_cvt_wt<<<gdim(wt), TPB, 0, stream>>>(Wtbf, Wm, Kd, Nc, Kpad, wt);
        int tiles = (Mpad / 16) * (Ncpad / 16);
        k_gemm_wmma<<<(tiles + 7) / 8, TPB, 0, stream>>>(Abf, Wtbf, C, M, Kpad, Nc, accum);
    };
    auto spmm = [&](const float* h, float* o, int F, float factor) {
        long tot = (long)E * ((F + 3) / 4);
        k_spmm<<<gdim(tot), TPB, 0, stream>>>(src, dst, nrm, h, o, E, F, factor);
    };

    // copy x into working buffer 0
    k_copy_scale<<<gdim((long)N * 128), TPB, 0, stream>>>(bufs[0], x, 1.f, (long)N * 128);

    struct LD { int Cin, Cout, K, gn, act; };
    const LD layers[6] = {
        {128, 190, 2, 0, 2}, {190, 256, 2, 1, 2}, {256, 169, 2, 2, 2},
        {169, 190, 5, -1, 1}, {190, 256, 1, -1, 1}, {256, 128, 3, 3, 2}};

    for (int L = 0; L < 6; ++L) {
        const LD& ld = layers[L];
        float* in  = bufs[0];
        float* out = bufs[1];
        float* t1  = bufs[2];
        float* t2  = bufs[3];
        long nin = (long)N * ld.Cin;

        // out = T0 @ W0
        gemm(in, convw[L], out, N, ld.Cin, ld.Cout, 0);
        if (ld.K > 1) {
            // T1 = prop(T0)
            k_zero<<<gdim(nin), TPB, 0, stream>>>(t1, nin);
            spmm(in, t1, ld.Cin, 1.f);
            gemm(t1, convw[L] + (long)ld.Cin * ld.Cout, out, N, ld.Cin, ld.Cout, 1);
            float* T0 = in; float* T1 = t1; float* T2 = t2;
            for (int k = 2; k < ld.K; ++k) {
                // T2 = 2*prop(T1) - T0  (init T2 = -T0, scatter with factor 2)
                k_copy_scale<<<gdim(nin), TPB, 0, stream>>>(T2, T0, -1.f, nin);
                spmm(T1, T2, ld.Cin, 2.f);
                gemm(T2, convw[L] + (long)k * ld.Cin * ld.Cout, out, N, ld.Cin, ld.Cout, 1);
                float* tmp = T0; T0 = T1; T1 = T2; T2 = tmp;
            }
        }
        long nout = (long)N * ld.Cout;
        if (ld.gn >= 0) {
            k_bias_act<<<gdim(nout), TPB, 0, stream>>>(out, convb[L], N, ld.Cout, 0);
            k_colmean<<<ld.Cout, TPB, 0, stream>>>(out, mbuf, N, ld.Cout);
            k_colvar<<<ld.Cout, TPB, 0, stream>>>(out, mbuf, bnms[ld.gn], vbuf, N, ld.Cout);
            k_gn_act<<<gdim(nout), TPB, 0, stream>>>(out, mbuf, bnms[ld.gn], vbuf,
                                                     bnw[ld.gn], bnb[ld.gn], N, ld.Cout, ld.act);
        } else {
            k_bias_act<<<gdim(nout), TPB, 0, stream>>>(out, convb[L], N, ld.Cout, ld.act);
        }
        float* tmp = bufs[0]; bufs[0] = bufs[1]; bufs[1] = tmp;
    }

    // ---- global mean pool + heads ----
    float* h = bufs[0]; // conv6 result, 128 channels
    k_zero<<<gdim((long)G * 128), TPB, 0, stream>>>(pooled, (long)G * 128);
    k_zero<<<gdim(G), TPB, 0, stream>>>(cnt, G);
    k_pool<<<gdim((long)N * 128), TPB, 0, stream>>>(h, batch, pooled, cnt, N, 128);
    k_pooldiv<<<gdim((long)G * 128), TPB, 0, stream>>>(pooled, cnt, G, 128);
    k_lin<<<gdim((long)G * 64), TPB, 0, stream>>>(pooled, lin1w, lin1b, l1out, G, 128, 64, 3);
    k_lin<<<gdim((long)G * 10), TPB, 0, stream>>>(l1out, lin2w, lin2b, out_f, G, 64, 10, 0);
}
